// Head_36575941493444
// MI455X (gfx1250) — matrix operand order repaired and checked by reading, not compiled, not run on hardware
//
#include <hip/hip_runtime.h>
#include <hip/hip_bf16.h>
#include <math.h>

// ---------------- WMMA types ----------------
typedef __attribute__((ext_vector_type(16))) _Float16 v16h;
typedef __attribute__((ext_vector_type(8)))  float    v8f;

// A-fragment (16x32 f16, MxK): per lane 8 halves at K0 and 8 halves at K0+16.
__device__ __forceinline__ v16h lds_frag_a(const _Float16* p) {
  union { v16h v; uint4 u[2]; } r;
  r.u[0] = *(const uint4*)(p);
  r.u[1] = *(const uint4*)(p + 16);
  return r.v;
}
// B-fragment (32x16 f16, KxN): per lane 8 K halves at K0 and 8 at K0+16 of one column
// (halves of a row of B^T stored row-major).
__device__ __forceinline__ v16h lds_frag_b(const _Float16* p) {
  union { v16h v; uint4 u[2]; } r;
  r.u[0] = *(const uint4*)(p);
  r.u[1] = *(const uint4*)(p + 16);
  return r.v;
}
// A-fragment built from global f32 x row (8 floats at c, 8 at c+16), cvt to f16.
__device__ __forceinline__ v16h gmem_frag_a_f32(const float* p) {
  float4 f0 = *(const float4*)(p);
  float4 f1 = *(const float4*)(p + 4);
  float4 f2 = *(const float4*)(p + 16);
  float4 f3 = *(const float4*)(p + 20);
  v16h a;
  a[0]=(_Float16)f0.x; a[1]=(_Float16)f0.y; a[2]=(_Float16)f0.z; a[3]=(_Float16)f0.w;
  a[4]=(_Float16)f1.x; a[5]=(_Float16)f1.y; a[6]=(_Float16)f1.z; a[7]=(_Float16)f1.w;
  a[8]=(_Float16)f2.x; a[9]=(_Float16)f2.y; a[10]=(_Float16)f2.z; a[11]=(_Float16)f2.w;
  a[12]=(_Float16)f3.x; a[13]=(_Float16)f3.y; a[14]=(_Float16)f3.z; a[15]=(_Float16)f3.w;
  return a;
}

#define WMMA_F16(A,B,C) __builtin_amdgcn_wmma_f32_16x16x32_f16(false,(A),false,(B),(short)0,(C),false,false)

// ---------------- LDS layout (162,816 bytes total) ----------------
// Bank-conflict-free strides for 16-lane fragment access:
//   Wt   stride 392 halves (784B = 196 dw, 196%64=4)
//   q/k  stride  72 halves (144B = 36 dw, starts all distinct)
//   wei/vT stride 136 halves (272B = 68 dw, 68%64=4)
#define WT_STRIDE   392   // each W^T: [48][392] f16 -> 37,632 B
#define QK_STRIDE   72    // q16/k16: [128][72] f16  -> 18,432 B each
#define W16_STRIDE  136   // wei16:  [128][136] f16  -> 34,816 B (aliases WtQ)
#define VT_STRIDE   136   // vT:     [48][136] f16   -> 13,056 B
#define OFF_WTQ 0
#define OFF_WTK 37632
#define OFF_WTV 75264
#define OFF_Q   112896
#define OFF_K   131328
#define OFF_VT  149760
#define LDS_BYTES 162816

#define T_DIM 128
#define C_DIM 384
#define H_DIM 48

// Stage W^T (f16) into LDS: W is [C][H] row-major, H=48 (mult of 4 -> float4
// chunks never cross a row). Transposed scatter into Wt[h][c].
__device__ __forceinline__ void stage_wt(const float* __restrict__ W,
                                         _Float16* __restrict__ Wt, int tid) {
  for (int i = tid; i < (C_DIM * H_DIM) / 4; i += 256) {
    int e = i * 4;
    int c = e / H_DIM, h = e - c * H_DIM;
    float4 f = *(const float4*)(W + e);
    Wt[(h + 0) * WT_STRIDE + c] = (_Float16)f.x;
    Wt[(h + 1) * WT_STRIDE + c] = (_Float16)f.y;
    Wt[(h + 2) * WT_STRIDE + c] = (_Float16)f.z;
    Wt[(h + 3) * WT_STRIDE + c] = (_Float16)f.w;
  }
}

__global__ __launch_bounds__(256)
void attn_head_kernel(const float* __restrict__ x,
                      const float* __restrict__ Wk,
                      const float* __restrict__ Wq,
                      const float* __restrict__ Wv,
                      float* __restrict__ out,
                      float scale) {
  __shared__ __align__(16) unsigned char smem[LDS_BYTES];
  _Float16* WtQ   = (_Float16*)(smem + OFF_WTQ);
  _Float16* WtK   = (_Float16*)(smem + OFF_WTK);
  _Float16* WtV   = (_Float16*)(smem + OFF_WTV);
  _Float16* wei16 = (_Float16*)(smem + OFF_WTQ);   // aliases WtQ (dead by phase 2)
  _Float16* q16   = (_Float16*)(smem + OFF_Q);
  _Float16* k16   = (_Float16*)(smem + OFF_K);
  _Float16* vT    = (_Float16*)(smem + OFF_VT);

  const int b    = blockIdx.x;
  const int tid  = threadIdx.x;
  const int wave = tid >> 5;
  const int lane = tid & 31;
  const int l15  = lane & 15;
  const int hl   = lane >> 4;          // 0 or 1
  const int mt   = wave * 16;          // this wave's M-tile base row

  // Zero K-pad columns 48..63 of q16/k16 (H=48 padded to K=64 for q.k^T).
  if (tid < 128) {
    uint4 z = {0u, 0u, 0u, 0u};
    *(uint4*)(q16 + tid * QK_STRIDE + 48) = z;
    *(uint4*)(q16 + tid * QK_STRIDE + 56) = z;
    *(uint4*)(k16 + tid * QK_STRIDE + 48) = z;
    *(uint4*)(k16 + tid * QK_STRIDE + 56) = z;
  }

  // ---- Phase 0: stage all three W^T into LDS ----
  stage_wt(Wq, WtQ, tid);
  stage_wt(Wk, WtK, tid);
  stage_wt(Wv, WtV, tid);
  __syncthreads();

  // ---- Phase 1: single pass over x -> q, k, v (9 accumulators/wave) ----
  const float* xrow = x + ((size_t)b * T_DIM + (mt + l15)) * C_DIM;
  v8f aq[3] = {}, ak[3] = {}, av[3] = {};
#pragma unroll
  for (int kk = 0; kk < 12; ++kk) {
    v16h a = gmem_frag_a_f32(xrow + kk * 32 + hl * 8);
    // Batch all 9 B-fragment loads, then the 9 WMMAs (lets scheduler overlap).
    v16h bq[3], bk[3], bv[3];
#pragma unroll
    for (int n = 0; n < 3; ++n) {
      const int off = (n * 16 + l15) * WT_STRIDE + kk * 32 + hl * 8;
      bq[n] = lds_frag_b(WtQ + off);
      bk[n] = lds_frag_b(WtK + off);
      bv[n] = lds_frag_b(WtV + off);
    }
#pragma unroll
    for (int n = 0; n < 3; ++n) aq[n] = WMMA_F16(a, bq[n], aq[n]);
#pragma unroll
    for (int n = 0; n < 3; ++n) ak[n] = WMMA_F16(a, bk[n], ak[n]);
#pragma unroll
    for (int n = 0; n < 3; ++n) av[n] = WMMA_F16(a, bv[n], av[n]);
  }

  // Scatter D fragments: element (t = mt+g+8*hl, h = n*16+l15).
#pragma unroll
  for (int n = 0; n < 3; ++n)
#pragma unroll
    for (int g = 0; g < 8; ++g) {
      const int t = mt + g + 8 * hl;
      const int h = n * 16 + l15;
      q16[t * QK_STRIDE + h] = (_Float16)aq[n][g];
      k16[t * QK_STRIDE + h] = (_Float16)ak[n][g];
      vT[h * VT_STRIDE + t]  = (_Float16)av[n][g];
    }
  __syncthreads();

  // ---- Phase 2: wei = softmax(mask(q.k^T * scale)), kept in registers ----
  // A = q row-major [t][K=64]; B = k row-major [s][K] == k^T in [N][K] layout.
  v8f w[8] = {};
  {
    const _Float16* qrow = q16 + (mt + l15) * QK_STRIDE;
#pragma unroll
    for (int kk = 0; kk < 2; ++kk) {
      v16h a = lds_frag_a(qrow + kk * 32 + hl * 8);
      v16h bf[8];
#pragma unroll
      for (int n = 0; n < 8; ++n)
        bf[n] = lds_frag_b(k16 + (n * 16 + l15) * QK_STRIDE + kk * 32 + hl * 8);
#pragma unroll
      for (int n = 0; n < 8; ++n) w[n] = WMMA_F16(a, bf[n], w[n]);
    }
  }
  // In-register row softmax. For VGPR g: row t = mt+g+8*hl, col = n*16+l15.
  // shfl_xor masks 1/2/4/8 stay within each 16-lane half -> per-row reduction.
#pragma unroll
  for (int g = 0; g < 8; ++g) {
    const int t = mt + g + 8 * hl;
    float vv[8];
    float m = -1e30f;
#pragma unroll
    for (int n = 0; n < 8; ++n) {
      float val = w[n][g] * scale;
      val = ((n * 16 + l15) <= t) ? val : -1e30f;   // causal mask
      vv[n] = val;
      m = fmaxf(m, val);
    }
    m = fmaxf(m, __shfl_xor(m, 1, 32));
    m = fmaxf(m, __shfl_xor(m, 2, 32));
    m = fmaxf(m, __shfl_xor(m, 4, 32));
    m = fmaxf(m, __shfl_xor(m, 8, 32));
    float s = 0.0f;
#pragma unroll
    for (int n = 0; n < 8; ++n) { vv[n] = __expf(vv[n] - m); s += vv[n]; }
    s += __shfl_xor(s, 1, 32);
    s += __shfl_xor(s, 2, 32);
    s += __shfl_xor(s, 4, 32);
    s += __shfl_xor(s, 8, 32);
    const float r = 1.0f / s;
#pragma unroll
    for (int n = 0; n < 8; ++n)
      wei16[t * W16_STRIDE + n * 16 + l15] = (_Float16)(vv[n] * r);
  }
  __syncthreads();

  // ---- Phase 3: out = wei @ v.  A = wei16 [t][s]; B = vT [h][s] ([N][K]). ----
  v8f o[3] = {};
  {
    const _Float16* wrow = wei16 + (mt + l15) * W16_STRIDE;
#pragma unroll
    for (int kk = 0; kk < 8; ++kk) {
      v16h a = lds_frag_a(wrow + kk * 32 + hl * 8);
      v16h bf[3];
#pragma unroll
      for (int n = 0; n < 3; ++n)
        bf[n] = lds_frag_b(vT + (n * 16 + l15) * VT_STRIDE + kk * 32 + hl * 8);
#pragma unroll
      for (int n = 0; n < 3; ++n) o[n] = WMMA_F16(a, bf[n], o[n]);
    }
  }
#pragma unroll
  for (int n = 0; n < 3; ++n)
#pragma unroll
    for (int g = 0; g < 8; ++g)
      out[((size_t)b * T_DIM + (mt + g + 8 * hl)) * H_DIM + n * 16 + l15] = o[n][g];
}

extern "C" void kernel_launch(void* const* d_in, const int* in_sizes, int n_in,
                              void* d_out, int out_size, void* d_ws, size_t ws_size,
                              hipStream_t stream) {
  const float* x  = (const float*)d_in[0];
  const float* Wk = (const float*)d_in[1];
  const float* Wq = (const float*)d_in[2];
  const float* Wv = (const float*)d_in[3];
  float* out = (float*)d_out;
  const int B = in_sizes[0] / (T_DIM * C_DIM);           // 2048
  const float scale = sqrtf((float)B);                   // reference bug: * sqrt(B)
  attn_head_kernel<<<dim3(B), dim3(256), 0, stream>>>(x, Wk, Wq, Wv, out, scale);
}
